// MultiHeadAttentionWithCache_42709154791566
// MI455X (gfx1250) — matrix operand order repaired and checked by reading, not compiled, not run on hardware
//
#include <hip/hip_runtime.h>
#include <hip/hip_bf16.h>

// MHA with KV cache for MI455X (gfx1250, wave32, WMMA f16->f32).
// Pipeline: f32->f16 casts, 3 NT WMMA GEMMs (Q,K,V with layout-permuting
// epilogues), flash-attention (base-2 online softmax, 16-row Q tiles per
// wave, 64-key blocks, P transposed via per-wave LDS), final NT GEMM -> f32.
// Workspace requirement: ~151 MB (150,994,944 bytes) of d_ws.

#define B_      2
#define S_      2048
#define DM      2048
#define H_      16
#define HD_     128
#define CACHED_ 2048
#define T_      4096
#define NEGV    (-1e9f)

typedef __attribute__((ext_vector_type(16))) _Float16 v16h;
typedef __attribute__((ext_vector_type(8)))  _Float16 v8h;
typedef __attribute__((ext_vector_type(4)))  _Float16 v4h;
typedef __attribute__((ext_vector_type(8)))  float    v8f;
typedef __attribute__((ext_vector_type(4)))  float    v4f;

static __device__ __forceinline__ v16h cat8(v8h a, v8h b) {
  return __builtin_shufflevector(a, b, 0,1,2,3,4,5,6,7,8,9,10,11,12,13,14,15);
}
static __device__ __forceinline__ v8f wmma16(v16h a, v16h b, v8f c) {
  // D = A(16x32 f16) * B(32x16 f16) + C(16x16 f32)
  return __builtin_amdgcn_wmma_f32_16x16x32_f16(false, a, false, b, (short)0, c,
                                                false, false);
}

// ---------------------------------------------------------------- converts
__global__ void cvt_f32_to_f16(const float* __restrict__ s,
                               _Float16* __restrict__ d, int n) {
  int stride = gridDim.x * blockDim.x * 4;
  for (int i = (blockIdx.x * blockDim.x + threadIdx.x) * 4; i < n; i += stride) {
    v4f x = *(const v4f*)(s + i);
    *(v4h*)(d + i) = __builtin_convertvector(x, v4h);
  }
}

// k_cache [B,H,CACHED,HD] f32 -> Kf rows [0,CACHED) of [B,H,T,HD] f16
__global__ void fill_k_cache(const float* __restrict__ src,
                             _Float16* __restrict__ dst) {
  size_t i = (size_t)blockIdx.x * blockDim.x + threadIdx.x;
  int dd = (int)(i & (HD_ - 1));
  int t  = (int)((i >> 7) & (CACHED_ - 1));
  int bh = (int)(i >> 18);
  dst[((size_t)bh * T_ + t) * HD_ + dd] = (_Float16)src[i];
}

// v_cache [B,H,CACHED,HD] f32 -> Vt[b,h,d,t] f16 (t in [0,CACHED))
__global__ void fill_v_cache_t(const float* __restrict__ src,
                               _Float16* __restrict__ dst) {
  size_t i = (size_t)blockIdx.x * blockDim.x + threadIdx.x;  // output-indexed
  int t  = (int)(i & (CACHED_ - 1));
  int dd = (int)((i >> 11) & (HD_ - 1));
  int bh = (int)(i >> 18);
  dst[((size_t)bh * HD_ + dd) * T_ + t] =
      (_Float16)src[((size_t)bh * CACHED_ + t) * HD_ + dd];
}

// ---------------------------------------------------------------- GEMM (NT)
// C[M,N] = A[M,K] * B[N,K]^T ; M=B_*S_=4096, N=DM, K=DM. Row-major f16 in.
// MODE 0: write f16 -> Q[B,H,S,HD]        (col n = h*HD+dd)
// MODE 1: write f16 -> Kf[B,H,T,HD]       rows CACHED+s
// MODE 2: write f16 -> Vt[B,H,HD,T]       cols CACHED+s
// MODE 3: write f32 -> Out[M,N] row-major
template <int MODE>
__global__ __launch_bounds__(256) void gemm_nt(const _Float16* __restrict__ A,
                                               const _Float16* __restrict__ Bw,
                                               _Float16* __restrict__ Ch,
                                               float* __restrict__ Cf) {
  const int N_ = DM, K_ = DM;
  const int lane = threadIdx.x & 31;
  const int wave = threadIdx.x >> 5;
  const int lo = lane & 15, hi = lane >> 4;
  const int m0 = blockIdx.x * 32 + (wave >> 2) * 16;
  const int n0 = blockIdx.y * 256 + (wave & 3) * 64;

  v8f c[4] = {v8f{}, v8f{}, v8f{}, v8f{}};
  const _Float16* arow = A + (size_t)(m0 + lo) * K_ + hi * 16;
  const _Float16* brow = Bw + (size_t)(n0 + lo) * K_ + hi * 16;

  for (int k0 = 0; k0 < K_; k0 += 32) {
    v8h alo = *(const v8h*)(arow + k0);
    v8h ahi = *(const v8h*)(arow + k0 + 8);
    v16h af = cat8(alo, ahi);
#pragma unroll
    for (int t = 0; t < 4; ++t) {
      v16h bf = *(const v16h*)(brow + (size_t)t * 16 * K_ + k0);
      c[t] = wmma16(af, bf, c[t]);
    }
  }

#pragma unroll
  for (int t = 0; t < 4; ++t) {
    const int nb = n0 + t * 16 + lo;
#pragma unroll
    for (int r = 0; r < 8; ++r) {
      const int row = m0 + hi * 8 + r;
      float v = c[t][r];
      if (MODE == 3) {
        Cf[(size_t)row * N_ + nb] = v;
      } else {
        const int b = row >> 11, s = row & (S_ - 1);     // S_=2048
        const int h = nb >> 7, dd = nb & (HD_ - 1);      // HD_=128
        size_t idx;
        if (MODE == 0)
          idx = (((size_t)(b * H_ + h)) * S_ + s) * HD_ + dd;
        else if (MODE == 1)
          idx = (((size_t)(b * H_ + h)) * T_ + (CACHED_ + s)) * HD_ + dd;
        else
          idx = (((size_t)(b * H_ + h)) * HD_ + dd) * T_ + (CACHED_ + s);
        Ch[idx] = (_Float16)v;
      }
    }
  }
}

// ---------------------------------------------------------------- attention
// One wave per 16-row Q tile; base-2 online softmax over 64-key blocks.
__global__ __launch_bounds__(256) void flash_attn_k(const _Float16* __restrict__ Q,
                                                    const _Float16* __restrict__ Kf,
                                                    const _Float16* __restrict__ Vt,
                                                    _Float16* __restrict__ At) {
  __shared__ _Float16 Pl[8][16][64];  // per-wave P-transpose scratch (16 KB)
  const int lane = threadIdx.x & 31;
  const int wave = threadIdx.x >> 5;
  const int lo = lane & 15, hi = lane >> 4;
  const int tile = blockIdx.x * 8 + wave;           // 0 .. B*H*(S/16)-1
  const int qt = tile & (S_ / 16 - 1);              // S_/16 = 128
  const int bh = tile >> 7;
  const int q0 = qt * 16;

  const _Float16* Qb = Q + ((size_t)bh * S_ + q0) * HD_;
  const _Float16* Kb = Kf + (size_t)bh * T_ * HD_;
  const _Float16* Vb = Vt + (size_t)bh * HD_ * T_;

  // Q A-fragments: 4 chunks of K=32 over HD
  v16h qf[4];
#pragma unroll
  for (int dch = 0; dch < 4; ++dch) {
    const _Float16* p = Qb + (size_t)lo * HD_ + dch * 32 + hi * 16;
    qf[dch] = cat8(*(const v8h*)p, *(const v8h*)(p + 8));
  }

  v8f o[8] = {v8f{}, v8f{}, v8f{}, v8f{}, v8f{}, v8f{}, v8f{}, v8f{}};
  float mrow[8], lrow[8];
#pragma unroll
  for (int r = 0; r < 8; ++r) { mrow[r] = -1e30f; lrow[r] = 0.f; }

  // scale folded with log2(e): p = 2^(s*scl2 - m), v_exp_f32 is base-2.
  const float scl2 = 0.12751744416679f;  // (1/sqrt(128)) * log2(e)
  const int kend = CACHED_ + q0 + 16;    // keys needed by this tile
  const int nblk = (kend + 63) >> 6;

  for (int kb = 0; kb < nblk; ++kb) {
    const int j0 = kb * 64;
    // ---- scores: S[16q x 64k] = Q * K^T (four 16-key C tiles)
    v8f s[4] = {v8f{}, v8f{}, v8f{}, v8f{}};
#pragma unroll
    for (int dch = 0; dch < 4; ++dch) {
#pragma unroll
      for (int j = 0; j < 4; ++j) {
        const _Float16* kp =
            Kb + (size_t)(j0 + j * 16 + lo) * HD_ + dch * 32 + hi * 16;
        s[j] = wmma16(qf[dch], *(const v16h*)kp, s[j]);
      }
    }
    if (j0 + 64 < kend) {  // prefetch next K block
      __builtin_prefetch((const void*)(Kb + (size_t)(j0 + 64 + lo) * HD_), 0, 1);
      __builtin_prefetch((const void*)(Kb + (size_t)(j0 + 96 + lo) * HD_), 0, 1);
    }

    // ---- base-2 online softmax (row = hi*8 + r across 16 lanes of a half)
    const int cbase = j0 + lo;
#pragma unroll
    for (int r = 0; r < 8; ++r) {
      const int row = hi * 8 + r;
      const int lim = CACHED_ + q0 + row;  // causal limit (inclusive)
      float v0 = (cbase      <= lim) ? s[0][r] * scl2 : NEGV;
      float v1 = (cbase + 16 <= lim) ? s[1][r] * scl2 : NEGV;
      float v2 = (cbase + 32 <= lim) ? s[2][r] * scl2 : NEGV;
      float v3 = (cbase + 48 <= lim) ? s[3][r] * scl2 : NEGV;
      float mx = fmaxf(fmaxf(v0, v1), fmaxf(v2, v3));
#pragma unroll
      for (int off = 8; off; off >>= 1) mx = fmaxf(mx, __shfl_xor(mx, off, 32));
      const float mnew = fmaxf(mrow[r], mx);
      const float alpha = exp2f(mrow[r] - mnew);
      const float p0 = exp2f(v0 - mnew);
      const float p1 = exp2f(v1 - mnew);
      const float p2 = exp2f(v2 - mnew);
      const float p3 = exp2f(v3 - mnew);
      float sum = (p0 + p1) + (p2 + p3);
#pragma unroll
      for (int off = 8; off; off >>= 1) sum += __shfl_xor(sum, off, 32);
      lrow[r] = lrow[r] * alpha + sum;
      mrow[r] = mnew;
      Pl[wave][row][lo]      = (_Float16)p0;  // C-layout -> LDS
      Pl[wave][row][16 + lo] = (_Float16)p1;
      Pl[wave][row][32 + lo] = (_Float16)p2;
      Pl[wave][row][48 + lo] = (_Float16)p3;
#pragma unroll
      for (int t = 0; t < 8; ++t) o[t][r] *= alpha;
    }

    // ---- P as two A-fragments (16x32 each) from LDS (same-wave: in-order)
    const _Float16* pr = &Pl[wave][lo][0];
    v16h pf0 = cat8(*(const v8h*)(pr + hi * 16), *(const v8h*)(pr + 8 + hi * 16));
    v16h pf1 = cat8(*(const v8h*)(pr + 32 + hi * 16), *(const v8h*)(pr + 40 + hi * 16));

    // ---- O += P * V  (Vt gives contiguous B-fragments)
#pragma unroll
    for (int t = 0; t < 8; ++t) {
      const _Float16* vp = Vb + (size_t)(t * 16 + lo) * T_ + j0 + hi * 16;
      o[t] = wmma16(pf0, *(const v16h*)vp, o[t]);
      o[t] = wmma16(pf1, *(const v16h*)(vp + 32), o[t]);
    }
  }

  // ---- finalize: divide by l, write At[B,S,H*HD] f16
  const int b = bh >> 4, h = bh & (H_ - 1);
#pragma unroll
  for (int r = 0; r < 8; ++r) {
    const float inv = 1.0f / lrow[r];
    const int srow = q0 + hi * 8 + r;
    _Float16* dst = At + ((size_t)(b * S_ + srow)) * DM + h * HD_;
#pragma unroll
    for (int t = 0; t < 8; ++t) dst[t * 16 + lo] = (_Float16)(o[t][r] * inv);
  }
}

// ---------------------------------------------------------------- launch
extern "C" void kernel_launch(void* const* d_in, const int* in_sizes, int n_in,
                              void* d_out, int out_size, void* d_ws, size_t ws_size,
                              hipStream_t stream) {
  const float* hs = (const float*)d_in[0];
  const float* Wq = (const float*)d_in[1];
  const float* Wk = (const float*)d_in[2];
  const float* Wv = (const float*)d_in[3];
  const float* Wo = (const float*)d_in[4];
  const float* kc = (const float*)d_in[5];
  const float* vc = (const float*)d_in[6];
  // d_in[7] (attention_mask) is reproduced analytically (causal).
  float* out = (float*)d_out;

  const size_t NX  = (size_t)B_ * S_ * DM;        // 8388608
  const size_t NW  = (size_t)DM * DM;             // 4194304
  const size_t NKV = (size_t)B_ * H_ * T_ * HD_;  // 16777216

  _Float16* ws  = (_Float16*)d_ws;
  _Float16* Xh  = ws;
  _Float16* Wqh = Xh + NX;
  _Float16* Wkh = Wqh + NW;
  _Float16* Wvh = Wkh + NW;
  _Float16* Woh = Wvh + NW;
  _Float16* Qb  = Woh + NW;
  _Float16* Kf  = Qb + NX;
  _Float16* Vt  = Kf + NKV;
  _Float16* At  = Vt + NKV;   // total 75497472 halves = ~151 MB

  cvt_f32_to_f16<<<2048, 256, 0, stream>>>(hs, Xh, (int)NX);
  cvt_f32_to_f16<<<2048, 256, 0, stream>>>(Wq, Wqh, (int)NW);
  cvt_f32_to_f16<<<2048, 256, 0, stream>>>(Wk, Wkh, (int)NW);
  cvt_f32_to_f16<<<2048, 256, 0, stream>>>(Wv, Wvh, (int)NW);
  cvt_f32_to_f16<<<2048, 256, 0, stream>>>(Wo, Woh, (int)NW);

  const int ncache = B_ * H_ * CACHED_ * HD_;  // 8388608
  fill_k_cache<<<ncache / 256, 256, 0, stream>>>(kc, Kf);
  fill_v_cache_t<<<ncache / 256, 256, 0, stream>>>(vc, Vt);

  dim3 gg((B_ * S_) / 32, DM / 256);  // (128, 8)
  gemm_nt<0><<<gg, 256, 0, stream>>>(Xh, Wqh, Qb, nullptr);
  gemm_nt<1><<<gg, 256, 0, stream>>>(Xh, Wkh, Kf, nullptr);
  gemm_nt<2><<<gg, 256, 0, stream>>>(Xh, Wvh, Vt, nullptr);

  flash_attn_k<<<(B_ * H_ * (S_ / 16)) / 8, 256, 0, stream>>>(Qb, Kf, Vt, At);

  gemm_nt<3><<<gg, 256, 0, stream>>>(At, Woh, nullptr, out);
}
